// Quantizer_45741401703160
// MI455X (gfx1250) — matrix operand order repaired and checked by reading, not compiled, not run on hardware
//
#include <hip/hip_runtime.h>
#include <hip/hip_bf16.h>
#include <stdint.h>

// ---------------------------------------------------------------------------
// VQ quantizer for MI455X (gfx1250, wave32, WMMA).
// Core: two 8192x8192x1024 distance GEMMs lowered to v_wmma_f32_16x16x32_bf16
// with fused per-row argmin (dist = |c|^2 - 2*x.c ; |x|^2 constant per row).
// Codebooks (16 MB bf16) are L2-resident (192 MB L2), so full-K streaming per
// workgroup costs L2 bandwidth, not HBM. A-tile (64 rows x 1024 bf16 = 128 KB)
// lives in LDS (320 KB / WGP).
// ---------------------------------------------------------------------------

typedef __attribute__((ext_vector_type(16))) __bf16 v16bf;
typedef __attribute__((ext_vector_type(8)))  float  v8f;

#define D_DIM   1024
#define K_DIM   8192
#define N_ROWS  8192
#define ROWS_WG 64
#define WAVES   8

static __device__ __forceinline__ uint16_t f2bf_rne(float f) {
    uint32_t u = __float_as_uint(f);
    uint32_t r = u + 0x7fffu + ((u >> 16) & 1u);
    return (uint16_t)(r >> 16);
}

// --- f32 -> bf16 conversion, optional row sum-of-squares ---------------------
__global__ __launch_bounds__(256)
void convert_norm_kernel(const float* __restrict__ src, uint16_t* __restrict__ dst,
                         float* __restrict__ norm) {
    const int row = blockIdx.x;
    const int tid = threadIdx.x;
    const float4 f = *(const float4*)(src + (size_t)row * D_DIM + tid * 4);
    uint32_t lo = (uint32_t)f2bf_rne(f.x) | ((uint32_t)f2bf_rne(f.y) << 16);
    uint32_t hi = (uint32_t)f2bf_rne(f.z) | ((uint32_t)f2bf_rne(f.w) << 16);
    uint2 pk; pk.x = lo; pk.y = hi;
    *(uint2*)(dst + (size_t)row * D_DIM + tid * 4) = pk;
    if (norm != nullptr) {
        __shared__ float sred[256];
        sred[tid] = f.x * f.x + f.y * f.y + f.z * f.z + f.w * f.w;
        __syncthreads();
        for (int o = 128; o > 0; o >>= 1) {
            if (tid < o) sred[tid] += sred[tid + o];
            __syncthreads();
        }
        if (tid == 0) norm[row] = sred[0];
    }
}

// --- bf16 row gather: dst[row] = src[idx[row]] -------------------------------
__global__ __launch_bounds__(256)
void gather_bf16_kernel(const uint16_t* __restrict__ src, const int* __restrict__ idx,
                        uint16_t* __restrict__ dst) {
    const int row = blockIdx.x;
    const int tid = threadIdx.x;
    const int s = idx[row];
    *(uint2*)(dst + (size_t)row * D_DIM + tid * 4) =
        *(const uint2*)(src + (size_t)s * D_DIM + tid * 4);
}

// --- fused distance-GEMM + argmin -------------------------------------------
// One workgroup: 64 rows x full K. 8 waves; wave w owns cols chunk*128+w*16.
// Per wave per chunk: 4 row-tiles of 16x16 f32 acc, 32 k-steps of
// v_wmma_f32_16x16x32_bf16 each.
__global__ __launch_bounds__(256)
void argmin_kernel(const uint16_t* __restrict__ A, const uint16_t* __restrict__ B,
                   const float* __restrict__ Bnorm, int* __restrict__ out_idx) {
    __shared__ __align__(16) uint16_t smA[ROWS_WG * D_DIM];   // 128 KB A-tile
    __shared__ float red_val[WAVES][ROWS_WG];
    __shared__ int   red_idx[WAVES][ROWS_WG];

    const int tid   = threadIdx.x;
    const int wave  = tid >> 5;
    const int lane  = tid & 31;
    const int hlane = lane & 15;       // col within tile / row within half
    const int half  = lane >> 4;
    const int rowbase = blockIdx.x * ROWS_WG;

    // Cooperative A-tile load: 64 rows x 2048 B = 128 KB, 16 B per transfer.
    {
        const uint4* src = (const uint4*)(A + (size_t)rowbase * D_DIM);
        uint4* dst = (uint4*)smA;
        const int total = ROWS_WG * D_DIM / 8;   // 8192 uint4
        for (int i = tid; i < total; i += 256) dst[i] = src[i];
    }
    __syncthreads();

    float mval[4][8];
    int   midx[4][8];
#pragma unroll
    for (int rt = 0; rt < 4; ++rt)
#pragma unroll
        for (int j = 0; j < 8; ++j) { mval[rt][j] = 3.4e38f; midx[rt][j] = 0; }

#pragma unroll 1
    for (int chunk = 0; chunk < K_DIM / 128; ++chunk) {
        const int col = chunk * 128 + wave * 16 + hlane;
        // B fragment (32x16 bf16): lane holds codebook row `col`,
        // 16 contiguous K-elements starting at d + half*16 -> one 32 B read.
        const uint16_t* bp = B + (size_t)col * D_DIM + half * 16;

        v8f acc[4] = {};
#pragma unroll 4
        for (int d = 0; d < D_DIM; d += 32) {
            const v16bf bfrag = *(const v16bf*)(bp + d);
#pragma unroll
            for (int rt = 0; rt < 4; ++rt) {
                // A fragment (16x32 bf16): lane holds row rt*16+hlane,
                // k = d + half*16 + {0..15} -> two 16 B LDS reads.
                const uint16_t* ap = smA + ((rt * 16 + hlane) * D_DIM) + d + half * 16;
                union { v16bf v; uint4 q[2]; } u;
                u.q[0] = *(const uint4*)(ap);
                u.q[1] = *(const uint4*)(ap + 8);
                acc[rt] = __builtin_amdgcn_wmma_f32_16x16x32_bf16(
                    false, u.v, false, bfrag, (short)0, acc[rt], false, false);
            }
        }

        const float cn = Bnorm[col];
#pragma unroll
        for (int rt = 0; rt < 4; ++rt)
#pragma unroll
            for (int j = 0; j < 8; ++j) {
                const float dist = cn - 2.0f * acc[rt][j];
                if (dist < mval[rt][j]) { mval[rt][j] = dist; midx[rt][j] = col; }
            }
    }

    // Min-reduce across the 16 lanes of each half (distinct columns, same row),
    // tie-break on lower index (matches argmin "first" semantics).
#pragma unroll
    for (int rt = 0; rt < 4; ++rt)
#pragma unroll
        for (int j = 0; j < 8; ++j) {
            float v = mval[rt][j];
            int  ix = midx[rt][j];
#pragma unroll
            for (int m = 1; m < 16; m <<= 1) {
                const float ov = __shfl_xor(v, m, 32);
                const int   oi = __shfl_xor(ix, m, 32);
                if (ov < v || (ov == v && oi < ix)) { v = ov; ix = oi; }
            }
            if (hlane == 0) {
                const int rl = rt * 16 + half * 8 + j;   // local row (C layout)
                red_val[wave][rl] = v;
                red_idx[wave][rl] = ix;
            }
        }
    __syncthreads();

    if (tid < ROWS_WG) {
        float best = red_val[0][tid];
        int   bix  = red_idx[0][tid];
#pragma unroll
        for (int w = 1; w < WAVES; ++w) {
            const float v = red_val[w][tid];
            const int  ix = red_idx[w][tid];
            if (v < best || (v == best && ix < bix)) { best = v; bix = ix; }
        }
        out_idx[rowbase + tid] = bix;
    }
}

// --- fused straight-through output + loss partials ---------------------------
static __device__ __forceinline__ double sq(float a) { return (double)a * (double)a; }

__global__ __launch_bounds__(256)
void out_loss_kernel(const float* __restrict__ flat, const float* __restrict__ incb,
                     const float* __restrict__ outcb, const int* __restrict__ xi,
                     const int* __restrict__ ri, float* __restrict__ out,
                     double* __restrict__ partial) {
    const int row = blockIdx.x;
    const int tid = threadIdx.x;
    const int a = xi[row];
    const int b = ri[row];
    const float4 f = *(const float4*)(flat  + (size_t)row * D_DIM + tid * 4);
    const float4 q = *(const float4*)(incb  + (size_t)a   * D_DIM + tid * 4);
    const float4 r = *(const float4*)(outcb + (size_t)b   * D_DIM + tid * 4);
    *(float4*)(out + (size_t)row * D_DIM + tid * 4) = r;  // out = rq (straight-through fwd)

    double s = 0.0;
    s += sq(f.x - r.x) + sq(f.x - q.x) + sq(q.x - r.x);
    s += sq(f.y - r.y) + sq(f.y - q.y) + sq(q.y - r.y);
    s += sq(f.z - r.z) + sq(f.z - q.z) + sq(q.z - r.z);
    s += sq(f.w - r.w) + sq(f.w - q.w) + sq(q.w - r.w);

    __shared__ double dred[256];
    dred[tid] = s;
    __syncthreads();
    for (int o = 128; o > 0; o >>= 1) {
        if (tid < o) dred[tid] += dred[tid + o];
        __syncthreads();
    }
    if (tid == 0) partial[row] = dred[0];   // deterministic per-row partial
}

__global__ __launch_bounds__(256)
void finalize_loss_kernel(const double* __restrict__ partial, float* __restrict__ loss_out) {
    __shared__ double dred[256];
    const int tid = threadIdx.x;
    double s = 0.0;
    for (int i = tid; i < N_ROWS; i += 256) s += partial[i];
    dred[tid] = s;
    __syncthreads();
    for (int o = 128; o > 0; o >>= 1) {
        if (tid < o) dred[tid] += dred[tid + o];
        __syncthreads();
    }
    if (tid == 0)
        *loss_out = (float)(1.25 * dred[0] / ((double)N_ROWS * (double)D_DIM));
}

// ---------------------------------------------------------------------------
extern "C" void kernel_launch(void* const* d_in, const int* in_sizes, int n_in,
                              void* d_out, int out_size, void* d_ws, size_t ws_size,
                              hipStream_t stream) {
    (void)in_sizes; (void)n_in; (void)ws_size;
    const float* x     = (const float*)d_in[0];   // [8192, 64, 16] == flat [8192,1024]
    const float* incb  = (const float*)d_in[1];   // [8192, 1024]
    const float* outcb = (const float*)d_in[2];   // [8192, 1024]
    float* out = (float*)d_out;

    // Workspace carve-out (~34 MB): two bf16 buffers are reused across phases.
    char* ws = (char*)d_ws;
    size_t off = 0;
    auto carve = [&](size_t bytes) -> char* {
        char* p = ws + off;
        off += (bytes + 255) & ~(size_t)255;
        return p;
    };
    uint16_t* bufA   = (uint16_t*)carve((size_t)N_ROWS * D_DIM * 2);  // flat_bf16 -> recon_bf16
    uint16_t* bufB   = (uint16_t*)carve((size_t)K_DIM  * D_DIM * 2);  // incb_bf16 -> outcb_bf16
    float*    nrm    = (float*)   carve((size_t)K_DIM * 4);           // |c|^2 (reused)
    int*      x_inds = (int*)     carve((size_t)N_ROWS * 4);
    int*      r_inds = (int*)     carve((size_t)N_ROWS * 4);
    double*   partial= (double*)  carve((size_t)N_ROWS * 8);

    // 1) flat -> bf16 (no norm needed; |x|^2 is constant per argmin row)
    convert_norm_kernel<<<N_ROWS, 256, 0, stream>>>(x, bufA, nullptr);
    // 2) in_codebook -> bf16 + |c|^2
    convert_norm_kernel<<<K_DIM, 256, 0, stream>>>(incb, bufB, nrm);
    // 3) x_inds = argmin_k |in_cb[k]|^2 - 2 flat . in_cb[k]
    argmin_kernel<<<N_ROWS / ROWS_WG, 256, 0, stream>>>(bufA, bufB, nrm, x_inds);
    // 4) recon_bf16 = in_cb_bf16[x_inds]  (reuses already-converted rows)
    gather_bf16_kernel<<<N_ROWS, 256, 0, stream>>>(bufB, x_inds, bufA);
    // 5) out_codebook -> bf16 + |c|^2  (overwrites incb_bf16 / nrm, now dead)
    convert_norm_kernel<<<K_DIM, 256, 0, stream>>>(outcb, bufB, nrm);
    // 6) r_inds = argmin_k |out_cb[k]|^2 - 2 recon . out_cb[k]
    argmin_kernel<<<N_ROWS / ROWS_WG, 256, 0, stream>>>(bufA, bufB, nrm, r_inds);
    // 7) out = out_cb[r_inds]; loss partials (deterministic per-row doubles)
    out_loss_kernel<<<N_ROWS, 256, 0, stream>>>(x, incb, outcb, x_inds, r_inds, out, partial);
    // 8) loss = 1.25 * (mse(f,r)+mse(f,q)+mse(q,r)), fixed-tree reduction
    finalize_loss_kernel<<<1, 256, 0, stream>>>(partial, out + (size_t)out_size - 1);
}
